// MultiHeadAttention_47674136985811
// MI455X (gfx1250) — matrix operand order repaired and checked by reading, not compiled, not run on hardware
//
#include <hip/hip_runtime.h>

typedef __bf16 bf16_t;
typedef __bf16 v16bf __attribute__((ext_vector_type(16)));
typedef float  v8f   __attribute__((ext_vector_type(8)));
typedef unsigned short u16;
typedef unsigned int   u32;

#define T_DIM 128
#define C_DIM 256
#define D_DIM 256
#define H_DIM 8
#define B_DIM 256
#define WELEMS (H_DIM * D_DIM * C_DIM)            /* 524288 bf16 elems per QKV weight tensor */
#define WTOT   (3 * WELEMS + C_DIM * C_DIM)       /* 1638400 u16: all weights */
#define XELEMS ((size_t)B_DIM * T_DIM * C_DIM)    /* 8388608 u16: x in bf16 */

// ---------- helpers ----------
__device__ __forceinline__ u16 bfbits(float f) {
  union { bf16_t b; u16 u; } c; c.b = (bf16_t)f; return c.u;  // RNE via v_cvt_*_bf16_f32
}

// A-operand (16x32 bf16): per-lane two 16B chunks at +0 and +16 elements
__device__ __forceinline__ v16bf ldA_bf(const u16* p) {
  union { v16bf v; uint4 q[2]; } u;
  u.q[0] = *(const uint4*)(p);
  u.q[1] = *(const uint4*)(p + 16);
  return u.v;
}
// B-operand (32x16 bf16): per-lane two 16B chunks at +0 and +16 elements
__device__ __forceinline__ v16bf ldB_bf(const u16* p) {
  union { v16bf v; uint4 q[2]; } u;
  u.q[0] = *(const uint4*)(p);
  u.q[1] = *(const uint4*)(p + 16);
  return u.v;
}
// A-operand built from global fp32 (fallback path when workspace too small)
__device__ __forceinline__ v16bf ldA_f32(const float* p) {
  float4 a = *(const float4*)(p);
  float4 b = *(const float4*)(p + 4);
  float4 c = *(const float4*)(p + 16);
  float4 d = *(const float4*)(p + 20);
  v16bf r;
  r[0]=(bf16_t)a.x;  r[1]=(bf16_t)a.y;  r[2]=(bf16_t)a.z;  r[3]=(bf16_t)a.w;
  r[4]=(bf16_t)b.x;  r[5]=(bf16_t)b.y;  r[6]=(bf16_t)b.z;  r[7]=(bf16_t)b.w;
  r[8]=(bf16_t)c.x;  r[9]=(bf16_t)c.y;  r[10]=(bf16_t)c.z; r[11]=(bf16_t)c.w;
  r[12]=(bf16_t)d.x; r[13]=(bf16_t)d.y; r[14]=(bf16_t)d.z; r[15]=(bf16_t)d.w;
  return r;
}

#define WMMA_BF16(A,B,C) \
  __builtin_amdgcn_wmma_f32_16x16x32_bf16(false,(A),false,(B),(short)0,(C),false,false)

// One 16-row x 256-col projection slab: dst = X(rows m0..m0+15) @ W^T + bias
// TRANS=false: dst[t*256+d] (row-major), TRANS=true: dst[d*128+t] (transposed, V^T)
template<bool TRANS, bool XBF>
__device__ __forceinline__ void proj16(const float* xrow_f, const u16* xrow_b,
                                       const u16* Wh, const float* biash, u16* dst,
                                       int m0, int ln, int lh) {
  v8f acc[16];
#pragma unroll
  for (int j = 0; j < 16; ++j) {
    float bb = biash[j * 16 + ln];
#pragma unroll
    for (int e = 0; e < 8; ++e) acc[j][e] = bb;
  }
#pragma unroll 1
  for (int kk = 0; kk < 8; ++kk) {
    int c0 = kk * 32;
    v16bf A = XBF ? ldA_bf(xrow_b + c0 + (lh ? 8 : 0))
                  : ldA_f32(xrow_f + c0 + (lh ? 8 : 0));
#pragma unroll
    for (int j = 0; j < 16; ++j) {
      v16bf Bm = ldB_bf(Wh + (size_t)(j * 16 + ln) * C_DIM + c0 + (lh ? 8 : 0));
      acc[j] = WMMA_BF16(A, Bm, acc[j]);
    }
  }
#pragma unroll
  for (int j = 0; j < 16; ++j) {
#pragma unroll
    for (int r = 0; r < 8; ++r) {
      int t = m0 + r + lh * 8;
      int d = j * 16 + ln;
      if (TRANS) dst[(size_t)d * T_DIM + t] = bfbits(acc[j][r]);
      else       dst[(size_t)t * D_DIM + d] = bfbits(acc[j][r]);
    }
  }
}

// ---------- fp32 -> bf16 pairwise converter (weights + x prologue) ----------
__global__ void cvt_pairs_kernel(const float* __restrict__ src,
                                 u16* __restrict__ dst, int npairs) {
  int i = blockIdx.x * 256 + threadIdx.x;
  if (i < npairs) {
    float2 f = ((const float2*)src)[i];
    union { bf16_t v[2]; u32 u; } c;
    c.v[0] = (bf16_t)f.x;
    c.v[1] = (bf16_t)f.y;
    ((u32*)dst)[i] = c.u;   // compiler emits v_cvt_pk_bf16_f32
  }
}

// ---------- fused MHA: one block per (head, batch) ----------
template<bool XBF>
__global__ __launch_bounds__(256, 4) void mha_fused_kernel(
    const float* __restrict__ x,   const u16* __restrict__ xbf,
    const float* __restrict__ bq,  const float* __restrict__ bk,
    const float* __restrict__ bv,  const float* __restrict__ bp,
    const u16*  __restrict__ wbf,
    float* __restrict__ out)
{
  __shared__ __align__(16) u16 Qs[T_DIM * D_DIM];   // 64KB: Q, then P (wave-local), then O (wave-local)
  __shared__ __align__(16) u16 KVs[T_DIM * D_DIM];  // 64KB: K, then V^T

  const int bx   = blockIdx.x;
  const int h    = bx >> 8;
  const int b    = bx & 255;
  const int lane = threadIdx.x & 31;
  const int wave = threadIdx.x >> 5;
  const int ln   = lane & 15;
  const int lh   = lane >> 4;
  const int m0   = wave << 4;

  const u16* WQ = wbf + (size_t)h * D_DIM * C_DIM;
  const u16* WK = wbf + (size_t)WELEMS     + (size_t)h * D_DIM * C_DIM;
  const u16* WV = wbf + (size_t)2 * WELEMS + (size_t)h * D_DIM * C_DIM;
  const u16* WP = wbf + (size_t)3 * WELEMS;

  const float* xrow_f = x   + ((size_t)b * T_DIM + m0 + ln) * C_DIM;
  const u16*   xrow_b = xbf + ((size_t)b * T_DIM + m0 + ln) * C_DIM;

  // ---- Phase 1: Q and K projections ----
  proj16<false, XBF>(xrow_f, xrow_b, WQ, bq + h * D_DIM, Qs,  m0, ln, lh);
  proj16<false, XBF>(xrow_f, xrow_b, WK, bk + h * D_DIM, KVs, m0, ln, lh);
  __syncthreads();

  // ---- Phase 2: att = softmax(mask(Q K^T / 16)) ----
  v8f accA[8];
#pragma unroll
  for (int j = 0; j < 8; ++j)
#pragma unroll
    for (int e = 0; e < 8; ++e) accA[j][e] = 0.f;

#pragma unroll 1
  for (int kk = 0; kk < 8; ++kk) {
    int c0 = kk * 32;
    v16bf A = ldA_bf(Qs + (size_t)(m0 + ln) * D_DIM + c0 + (lh ? 8 : 0));
#pragma unroll
    for (int j = 0; j < 8; ++j) {
      v16bf Bm = ldB_bf(KVs + (size_t)(j * 16 + ln) * D_DIM + c0 + (lh ? 8 : 0));
      accA[j] = WMMA_BF16(A, Bm, accA[j]);
    }
  }
  const float scale = 0.0625f;   // C^-0.5 = 256^-0.5
#pragma unroll
  for (int j = 0; j < 8; ++j)
#pragma unroll
    for (int r = 0; r < 8; ++r) {
      int t = m0 + r + lh * 8;
      int s = j * 16 + ln;
      float v = accA[j][r] * scale;
      accA[j][r] = (s <= t) ? v : -__builtin_inff();
    }
  // row-wise softmax: each C-tile row lives in one 16-lane half
#pragma unroll
  for (int r = 0; r < 8; ++r) {
    float mx = accA[0][r];
#pragma unroll
    for (int j = 1; j < 8; ++j) mx = fmaxf(mx, accA[j][r]);
#pragma unroll
    for (int off = 1; off < 16; off <<= 1) mx = fmaxf(mx, __shfl_xor(mx, off, 32));
    float sum = 0.f;
#pragma unroll
    for (int j = 0; j < 8; ++j) { float e = __expf(accA[j][r] - mx); accA[j][r] = e; sum += e; }
#pragma unroll
    for (int off = 1; off < 16; off <<= 1) sum += __shfl_xor(sum, off, 32);
    float inv = 1.0f / sum;
#pragma unroll
    for (int j = 0; j < 8; ++j) accA[j][r] *= inv;
  }
  // P (16x128 bf16) into this wave's own Qs region (its Q rows are dead now)
  u16* Ps = Qs + (size_t)m0 * D_DIM;
#pragma unroll
  for (int j = 0; j < 8; ++j)
#pragma unroll
    for (int r = 0; r < 8; ++r)
      Ps[(size_t)(r + lh * 8) * T_DIM + j * 16 + ln] = bfbits(accA[j][r]);
  __syncthreads();   // everyone done reading K before V^T overwrites KVs

  // ---- Phase 3: V projection, stored transposed V^T[d][s] ----
  proj16<true, XBF>(xrow_f, xrow_b, WV, bv + h * D_DIM, KVs, m0, ln, lh);
  __syncthreads();

  // ---- Phase 4: O = P @ V ----
  v8f accO[16];
#pragma unroll
  for (int j = 0; j < 16; ++j)
#pragma unroll
    for (int e = 0; e < 8; ++e) accO[j][e] = 0.f;

#pragma unroll 1
  for (int kk = 0; kk < 4; ++kk) {
    int s0 = kk * 32;
    v16bf A = ldA_bf(Ps + (size_t)ln * T_DIM + s0 + (lh ? 8 : 0));
#pragma unroll
    for (int j = 0; j < 16; ++j) {
      v16bf Bm = ldB_bf(KVs + (size_t)(j * 16 + ln) * T_DIM + s0 + (lh ? 8 : 0));
      accO[j] = WMMA_BF16(A, Bm, accO[j]);
    }
  }
  // O (16x256 bf16) into wave-local Qs region (P dead) — no block barrier needed
  u16* Os = Qs + (size_t)m0 * D_DIM;
#pragma unroll
  for (int j = 0; j < 16; ++j)
#pragma unroll
    for (int r = 0; r < 8; ++r)
      Os[(size_t)(r + lh * 8) * D_DIM + j * 16 + ln] = bfbits(accO[j][r]);
  asm volatile("" ::: "memory");   // keep store->load program order at IR level

  // ---- Phase 5: out = O @ Wp^T + bp ----
  v8f accF[16];
#pragma unroll
  for (int j = 0; j < 16; ++j) {
    float bb = bp[j * 16 + ln];
#pragma unroll
    for (int e = 0; e < 8; ++e) accF[j][e] = bb;
  }
#pragma unroll 1
  for (int kk = 0; kk < 8; ++kk) {
    int c0 = kk * 32;
    v16bf A = ldA_bf(Os + (size_t)ln * D_DIM + c0 + (lh ? 8 : 0));
#pragma unroll
    for (int j = 0; j < 16; ++j) {
      v16bf Bm = ldB_bf(WP + (size_t)(j * 16 + ln) * D_DIM + c0 + (lh ? 8 : 0));
      accF[j] = WMMA_BF16(A, Bm, accF[j]);
    }
  }
  float* outb = out + (size_t)(h * B_DIM + b) * T_DIM * C_DIM;
#pragma unroll
  for (int j = 0; j < 16; ++j)
#pragma unroll
    for (int r = 0; r < 8; ++r) {
      int t = m0 + r + lh * 8;
      outb[(size_t)t * C_DIM + j * 16 + ln] = accF[j][r];
    }
}

extern "C" void kernel_launch(void* const* d_in, const int* in_sizes, int n_in,
                              void* d_out, int out_size, void* d_ws, size_t ws_size,
                              hipStream_t stream) {
  const float* x  = (const float*)d_in[0];
  const float* Wq = (const float*)d_in[1];
  const float* bq = (const float*)d_in[2];
  const float* Wk = (const float*)d_in[3];
  const float* bk = (const float*)d_in[4];
  const float* Wv = (const float*)d_in[5];
  const float* bv = (const float*)d_in[6];
  const float* Wp = (const float*)d_in[7];
  const float* bp = (const float*)d_in[8];
  float* out = (float*)d_out;

  u16* wbf = (u16*)d_ws;          // weights: WTOT u16 (~3.3 MB)
  u16* xbf = wbf + WTOT;          // x in bf16: XELEMS u16 (~16.8 MB)
  const bool use_xbf = ws_size >= (size_t)(WTOT + XELEMS) * sizeof(u16);

  // weights -> bf16 (pairwise; all sizes are even)
  cvt_pairs_kernel<<<dim3(WELEMS/2/256), dim3(256), 0, stream>>>(Wq, wbf, WELEMS/2);
  cvt_pairs_kernel<<<dim3(WELEMS/2/256), dim3(256), 0, stream>>>(Wk, wbf + WELEMS, WELEMS/2);
  cvt_pairs_kernel<<<dim3(WELEMS/2/256), dim3(256), 0, stream>>>(Wv, wbf + 2*WELEMS, WELEMS/2);
  cvt_pairs_kernel<<<dim3(C_DIM*C_DIM/2/256), dim3(256), 0, stream>>>(Wp, wbf + 3*WELEMS, C_DIM*C_DIM/2);

  if (use_xbf) {
    cvt_pairs_kernel<<<dim3((int)(XELEMS/2/256)), dim3(256), 0, stream>>>(x, xbf, (int)(XELEMS/2));
    mha_fused_kernel<true><<<dim3(H_DIM * B_DIM), dim3(256), 0, stream>>>(
        x, xbf, bq, bk, bv, bp, wbf, out);
  } else {
    mha_fused_kernel<false><<<dim3(H_DIM * B_DIM), dim3(256), 0, stream>>>(
        x, xbf, bq, bk, bv, bp, wbf, out);
  }
}
